// SparseUvuTensorProduct_86835648790773
// MI455X (gfx1250) — matrix operand order repaired and checked by reading, not compiled, not run on hardware
//
#include <hip/hip_runtime.h>
#include <cmath>
#include <complex>
#include <cstring>
#include <algorithm>

// ---------------- problem constants (static from the reference) ----------------
constexpr int DIM1  = 1152;   // 128*(1+3+5)
constexpr int DIM2  = 16;     // 1+3+5+7
constexpr int DIMO  = 3968;   // 128*(1+3+3+5+5+7+7)
constexpr int NPATH = 24;
constexpr int NK    = 96;     // padded K dim: segments grouped by x1 component
constexpr int WSTR  = 104;    // LDS row stride (bf16 elems): 16B-aligned, bank-friendly
constexpr int XPS   = 12;     // x1 dup-pair row stride (dwords), 48B = 16B-aligned
constexpr int WAVES = 4;      // one sample per wave
constexpr int BLOCK = WAVES * 32;

// LDS layout (bytes)
constexpr int WT_OFF = 0;                              // bf16 [128][WSTR] (per WG)
constexpr int SK_OFF = WT_OFF + 128 * WSTR * 2;        // bf16 [WAVES][32][WSTR]
constexpr int XP_OFF = SK_OFF + WAVES * 32 * WSTR * 2; // u32  [WAVES][128][XPS]
constexpr int SO_OFF = XP_OFF + WAVES * 128 * XPS * 4; // f32  [WAVES][DIMO]
constexpr int SX_OFF = SO_OFF + WAVES * DIMO * 4;      // f32  [WAVES][16]
constexpr int SMEM_TOTAL = SX_OFF + WAVES * 16 * 4;    // = 141568 B

// workspace layout (bytes)
constexpr int MAXC = 2700;
constexpr int MAXE = 4096;
constexpr int WS_CELLS = 0;
constexpr int WS_ENTS  = WS_CELLS + MAXC * 16;
constexpr size_t WS_TOTAL = WS_ENTS + (size_t)MAXE * 8;

struct Cell { int k, n, off, cnt; };          // K-matrix nonzero cell
struct __align__(8) Ent { float v; int d; };  // one pw*cg value, x2 component d

// k ordering: segments grouped by x1 component g (pair- and even-aligned):
//   g0: k0-3 = l1=0 paths p0-3
//   g1-3: k4-33   (10 each) = l1=1 paths p4-12 + pad, a = g-1
//   g4-8: k34-93  (12 each) = l1=2 paths p13-23 + pad, a = g-4
//   k94-95: pad
// Fragment pair (ks, j) covers k = ks*32 + kh*8 + {0,2,4,6,16,18,20,22}[j];
// its x1 factor is xp[g] (kh-invariant) or y[g] = kh ? xp[g+1] : xp[g].
__constant__ int c_kpath[NK] = {
    0,1,2,3,
    4,5,6,7,8,9,10,11,12,24,  4,5,6,7,8,9,10,11,12,24,  4,5,6,7,8,9,10,11,12,24,
    13,14,15,16,17,18,19,20,21,22,23,24,  13,14,15,16,17,18,19,20,21,22,23,24,
    13,14,15,16,17,18,19,20,21,22,23,24,  13,14,15,16,17,18,19,20,21,22,23,24,
    13,14,15,16,17,18,19,20,21,22,23,24,  24,24};
// x1 flat column (u=0) and per-channel stride for each g
constexpr int G_BASE[9] = {0, 128, 129, 130, 512, 513, 514, 515, 516};
constexpr int G_STR[9]  = {1, 3, 3, 3, 5, 5, 5, 5, 5};

// output column tables: flat = obase[n] + u*odim[n]  (n = 0..30, 31 = pad)
__constant__ int c_obase[32] = {
    0, 128, 129, 130, 512, 513, 514, 896, 897, 898, 899, 900,
    1536, 1537, 1538, 1539, 1540, 2176, 2177, 2178, 2179, 2180, 2181, 2182,
    3072, 3073, 3074, 3075, 3076, 3077, 3078, 0};
__constant__ int c_odim[32] = {
    1, 3,3,3, 3,3,3, 5,5,5,5,5, 5,5,5,5,5,
    7,7,7,7,7,7,7, 7,7,7,7,7,7,7, 1};

// ---------------- WMMA fragment types ----------------
typedef __attribute__((ext_vector_type(2)))  __bf16 bf16x2;
typedef __attribute__((ext_vector_type(8)))  __bf16 bf16x8;
typedef __attribute__((ext_vector_type(16))) __bf16 bf16x16;
typedef __attribute__((ext_vector_type(8)))  float  v8f;
union Frag { bf16x16 v; bf16x8 h[2]; unsigned int u[8]; };

__device__ inline unsigned int pkmul(unsigned int a, unsigned int b) {
  union PK { unsigned int u; bf16x2 v; } x, y, z;
  x.u = a; y.u = b;
  z.v = x.v * y.v;                       // v_pk_mul_bf16
  return z.u;
}

// ---------------- device kernel ----------------
__global__ __launch_bounds__(BLOCK, 1) void tp_uvu_wmma_kernel(
    const float* __restrict__ x1, const float* __restrict__ x2,
    const float* __restrict__ wgt, float* __restrict__ out,
    const int4* __restrict__ gCells, int nCells,
    const Ent*  __restrict__ gEnts, int nSamples)
{
  extern __shared__ char smem[];
  __bf16*       sWT = (__bf16*)(smem + WT_OFF);
  __bf16*       sK  = (__bf16*)(smem + SK_OFF);
  unsigned int* sXP = (unsigned int*)(smem + XP_OFF);
  float*        sO  = (float*)(smem + SO_OFF);
  float*        sX  = (float*)(smem + SX_OFF);

  const int tid  = threadIdx.x;
  const int lane = tid & 31;
  const int wave = tid >> 5;

  // ---- once per WG: weight table in WMMA-A fragment order (bf16) ----
  for (int k = 0; k < NK; ++k) {
    int p = c_kpath[k];
    float wv = (p < NPATH) ? wgt[p * 128 + tid] : 0.0f;
    sWT[tid * WSTR + k] = (__bf16)wv;
  }
  __syncthreads();

  __bf16*       myK  = sK + wave * 32 * WSTR;
  unsigned int* myXP = sXP + wave * 128 * XPS;
  float*        myO  = sO + wave * DIMO;
  float*        myX  = sX + wave * 16;

  const int n   = lane & 15;
  const int kh  = lane >> 4;
  const bool khi = (kh != 0);

  const int step = gridDim.x * WAVES;
  for (int b0 = blockIdx.x * WAVES; b0 < nSamples; b0 += step) {
    const int b = b0 + wave;                 // uniform within the wave
    if (b >= nSamples) continue;             // whole-wave uniform

    // ---- per-sample staging (all same-wave LDS: hardware keeps it in order) --
    if (lane < 16) myX[lane] = x2[(size_t)b * DIM2 + lane];
    {
      int4 z = make_int4(0, 0, 0, 0);
      int4* kz = (int4*)myK;
      for (int i = lane; i < (32 * WSTR * 2) / 16; i += 32) kz[i] = z;
    }
    if (b + step < nSamples)
      __builtin_prefetch(x1 + (size_t)(b + step) * DIM1, 0, 1);

    // x1 -> duplicated bf16 pairs, one dword per (u, g)
    const float* x1b = x1 + (size_t)b * DIM1;
#pragma unroll
    for (int g = 0; g < 9; ++g) {
#pragma unroll
      for (int j = 0; j < 4; ++j) {
        int u = lane + 32 * j;
        union { __bf16 h; unsigned short s; } cv;
        cv.h = (__bf16)x1b[G_BASE[g] + u * G_STR[g]];
        myXP[u * XPS + g] = (unsigned int)cv.s * 0x10001u;
      }
    }

    // ---- K build: Kt[n][k] = pw * sum_d cg[a,d,c] * x2[d] (bf16) ----
    for (int ci = lane; ci < nCells; ci += 32) {
      int4 cl = gCells[ci];                  // {k, n, off, cnt}
      float acc = 0.0f;
      for (int e = 0; e < cl.w; ++e) {
        Ent en = gEnts[cl.z + e];
        acc += en.v * myX[en.d];
      }
      myK[cl.y * WSTR + cl.x] = (__bf16)acc;
    }

    // ---- B fragments (K matrix) ----
    Frag Bf[2][3];
#pragma unroll
    for (int nt = 0; nt < 2; ++nt)
#pragma unroll
      for (int ks = 0; ks < 3; ++ks) {
        const __bf16* p = myK + (nt * 16 + n) * WSTR + ks * 32 + kh * 8;
        Bf[nt][ks].h[0] = *(const bf16x8*)(p);
        Bf[nt][ks].h[1] = *(const bf16x8*)(p + 16);
      }

    // output column info (hoisted; nt=1,n=15 is the pad column)
    const int ob0 = c_obase[n],      od0 = c_odim[n];
    const int ob1 = c_obase[16 + n], od1 = c_odim[16 + n];
    const bool v1 = (n < 15);

    // ---- GEMM: A fragments built in registers from Wtab x broadcast x1 ----
#pragma unroll 1
    for (int mt = 0; mt < 8; ++mt) {
      const int u = mt * 16 + n;
      // x1 dup-pairs for this row: explicit named scalars (no register arrays!)
      const unsigned int* xpr = myXP + u * XPS;
      const uint4 xa = *(const uint4*)(xpr);
      const uint4 xb = *(const uint4*)(xpr + 4);
      const unsigned int x0v = xa.x, x1v = xa.y, x2v = xa.z, x3v = xa.w;
      const unsigned int x4v = xb.x, x5v = xb.y, x6v = xb.z, x7v = xb.w;
      const unsigned int x8v = xpr[8];
      // kh-adjusted values: y[g] = kh ? xp[g+1] : xp[g]  (8 value-domain selects)
      const unsigned int y0 = khi ? x1v : x0v;
      const unsigned int y1 = khi ? x2v : x1v;
      const unsigned int y2 = khi ? x3v : x2v;
      const unsigned int y3 = khi ? x4v : x3v;
      const unsigned int y4 = khi ? x5v : x4v;
      const unsigned int y5 = khi ? x6v : x5v;
      const unsigned int y6 = khi ? x7v : x6v;
      const unsigned int y7 = khi ? x8v : x7v;

      const __bf16* wrow = sWT + u * WSTR;
      const __bf16* wp0 = wrow + kh * 8;
      const __bf16* wp1 = wrow + 32 + kh * 8;
      const __bf16* wp2 = wrow + 64 + kh * 8;
      const uint4 w00 = *(const uint4*)(wp0);
      const uint4 w01 = *(const uint4*)(wp0 + 16);
      const uint4 w10 = *(const uint4*)(wp1);
      const uint4 w11 = *(const uint4*)(wp1 + 16);
      const uint4 w20 = *(const uint4*)(wp2);
      const uint4 w21 = *(const uint4*)(wp2 + 16);

      Frag Af[3];
      // ks = 0  (k 0..31: g segments 0 | 1 | 2 | 3)
      Af[0].u[0] = pkmul(y0,  w00.x);
      Af[0].u[1] = pkmul(y0,  w00.y);
      Af[0].u[2] = pkmul(x1v, w00.z);
      Af[0].u[3] = pkmul(y1,  w00.w);
      Af[0].u[4] = pkmul(y2,  w01.x);
      Af[0].u[5] = pkmul(y2,  w01.y);
      Af[0].u[6] = pkmul(y2,  w01.z);
      Af[0].u[7] = pkmul(y2,  w01.w);
      // ks = 1  (k 32..63: g segments 3 | 4 | 5 | 6)
      Af[1].u[0] = pkmul(y3,  w10.x);
      Af[1].u[1] = pkmul(x4v, w10.y);
      Af[1].u[2] = pkmul(x4v, w10.z);
      Af[1].u[3] = pkmul(y4,  w10.w);
      Af[1].u[4] = pkmul(x5v, w11.x);
      Af[1].u[5] = pkmul(y5,  w11.y);
      Af[1].u[6] = pkmul(y5,  w11.z);
      Af[1].u[7] = pkmul(y5,  w11.w);
      // ks = 2  (k 64..95: g segments 6 | 7 | 8 | 8)
      Af[2].u[0] = pkmul(y6,  w20.x);
      Af[2].u[1] = pkmul(y6,  w20.y);
      Af[2].u[2] = pkmul(y6,  w20.z);
      Af[2].u[3] = pkmul(x7v, w20.w);
      Af[2].u[4] = pkmul(y7,  w21.x);
      Af[2].u[5] = pkmul(x8v, w21.y);
      Af[2].u[6] = pkmul(x8v, w21.z);
      Af[2].u[7] = pkmul(x8v, w21.w);

#pragma unroll
      for (int nt = 0; nt < 2; ++nt) {
        v8f acc = {0.f, 0.f, 0.f, 0.f, 0.f, 0.f, 0.f, 0.f};
#pragma unroll
        for (int ks = 0; ks < 3; ++ks)
          acc = __builtin_amdgcn_wmma_f32_16x16x32_bf16(
              false, Af[ks].v, false, Bf[nt][ks].v, (short)0, acc,
              false, false);
        if (nt == 0 || v1) {                   // skip pad column
          const int ob = nt ? ob1 : ob0;
          const int od = nt ? od1 : od0;
          int off = ob + (mt * 16 + kh * 8) * od;   // 32-bit LDS addressing
#pragma unroll
          for (int r = 0; r < 8; ++r) { myO[off] = acc[r]; off += od; }
        }
      }
    }

    // ---- stream staged row to HBM: 31 coalesced b128 stores ----
    float* dst = out + (size_t)b * DIMO;
#pragma unroll 1
    for (int i = 0; i < DIMO / 128; ++i) {
      const int e = (i * 32 + lane) * 4;
      float4 v = *(const float4*)(myO + e);
      *(float4*)(dst + e) = v;
    }
  }
}

// ---------------- host: exact CG tables (same math as the reference) ----------------
static double fct(int n) { double r = 1.0; for (int i = 2; i <= n; ++i) r *= i; return r; }

static double su2cg(int j1, int m1, int j2, int m2, int j3, int m3) {
  if (m1 + m2 != m3) return 0.0;
  int vmin = std::max(std::max(-j1 + j2 + m3, -j1 + m1), 0);
  int vmax = std::min(std::min(j2 + j3 + m1, j3 - j1 + j2), j3 + m3);
  if (vmax < vmin) return 0.0;
  double C = (2.0 * j3 + 1.0) *
      (fct(j3 + j1 - j2) * fct(j3 - j1 + j2) * fct(j1 + j2 - j3) *
       fct(j3 + m3) * fct(j3 - m3)) /
      (fct(j1 + j2 + j3 + 1) * fct(j1 - m1) * fct(j1 + m1) *
       fct(j2 - m2) * fct(j2 + m2));
  double S = 0.0;
  for (int v = vmin; v <= vmax; ++v) {
    double t = (fct(j2 + j3 + m1 - v) * fct(j1 - m1 + v)) /
               (fct(v) * fct(j3 - j1 + j2 - v) * fct(j3 + m3 - v) *
                fct(v + j1 - j2 - m3));
    S += ((v + j2 + m2) & 1) ? -t : t;
  }
  return S * std::sqrt(C);
}

typedef std::complex<double> cd;
static void qmat(int l, cd q[7][7]) {
  for (int i = 0; i < 7; ++i) for (int j = 0; j < 7; ++j) q[i][j] = 0.0;
  const double s = 1.0 / std::sqrt(2.0);
  for (int m = -l; m < 0; ++m) { q[l + m][l - m] = s; q[l + m][l + m] = cd(0.0, -s); }
  q[l][l] = 1.0;
  for (int m = 1; m <= l; ++m) {
    double sg = (m & 1) ? -1.0 : 1.0;
    q[l + m][l + m] = sg * s; q[l + m][l - m] = cd(0.0, sg * s);
  }
  cd ph(1.0, 0.0);
  switch (l & 3) { case 1: ph = cd(0.0, -1.0); break;
                   case 2: ph = cd(-1.0, 0.0); break;
                   case 3: ph = cd(0.0, 1.0); break; default: break; }
  for (int i = 0; i < 2 * l + 1; ++i)
    for (int j = 0; j < 2 * l + 1; ++j) q[i][j] *= ph;
}

static void wig3j(int l1, int l2, int l3, double w[7][7][7]) {
  int d1 = 2 * l1 + 1, d2 = 2 * l2 + 1, d3 = 2 * l3 + 1;
  double Cs[7][7][7]; std::memset(Cs, 0, sizeof(Cs));
  for (int m1 = -l1; m1 <= l1; ++m1)
    for (int m2 = -l2; m2 <= l2; ++m2) {
      int m3 = m1 + m2;
      if (m3 < -l3 || m3 > l3) continue;
      Cs[l1 + m1][l2 + m2][l3 + m3] = su2cg(l1, m1, l2, m2, l3, m3);
    }
  cd q1[7][7], q2[7][7], q3[7][7];
  qmat(l1, q1); qmat(l2, q2); qmat(l3, q3);
  double nrm = 0.0;
  for (int j = 0; j < d1; ++j)
    for (int t = 0; t < d2; ++t)
      for (int m = 0; m < d3; ++m) {
        cd a(0.0, 0.0);
        for (int i = 0; i < d1; ++i)
          for (int k = 0; k < d2; ++k) {
            cd pq = q1[i][j] * q2[k][t];
            for (int nn = 0; nn < d3; ++nn)
              a += pq * std::conj(q3[nn][m]) * Cs[i][k][nn];
          }
        w[j][t][m] = a.real();
        nrm += a.real() * a.real();
      }
  nrm = std::sqrt(nrm); if (nrm < 1e-30) nrm = 1.0;
  for (int j = 0; j < d1; ++j)
    for (int t = 0; t < d2; ++t)
      for (int m = 0; m < d3; ++m) w[j][t][m] /= nrm;
}

// path enumeration in the reference's _build_paths() order: (l1, l2, io)
static const int g_pd[NPATH][3] = {
    {0,0,0},{0,1,1},{0,2,3},{0,3,5},
    {1,0,1},{1,1,0},{1,1,2},{1,1,3},{1,2,1},{1,2,4},{1,2,5},{1,3,3},{1,3,6},
    {2,0,3},{2,1,1},{2,1,4},{2,1,5},{2,2,0},{2,2,2},{2,2,3},{2,2,6},
    {2,3,1},{2,3,4},{2,3,5}};

static inline int kOfPathA(int p, int a) {
  if (p < 4)  return p;                       // l1=0, a=0
  if (p < 13) return 4 + a * 10 + (p - 4);    // l1=1 segments (len 10)
  return 34 + a * 12 + (p - 13);              // l1=2 segments (len 12)
}

static void build_tables(Cell* cells, int& nC, Ent* ents, int& nE) {
  static const int ioL[7]   = {0, 1, 1, 2, 2, 3, 3};
  static const int xcnt[7]  = {3, 5, 2, 5, 3, 4, 2};
  static const int nOffT[7] = {0, 1, 4, 7, 12, 17, 24};
  nC = 0; nE = 0;
  for (int p = 0; p < NPATH; ++p) {
    int l1 = g_pd[p][0], l2 = g_pd[p][1], io = g_pd[p][2], lo = ioL[io];
    int d1 = 2 * l1 + 1, d2 = 2 * l2 + 1, dd = 2 * lo + 1;
    double w3[7][7][7]; wig3j(l1, l2, lo, w3);
    double pw = std::sqrt((2.0 * lo + 1.0) / (double)xcnt[io]);
    for (int a = 0; a < d1; ++a)
      for (int c = 0; c < dd; ++c) {
        int off = nE, cnt = 0;
        for (int d = 0; d < d2; ++d) {
          double v = w3[a][d][c];
          if (std::fabs(v) > 1e-10 && nE < MAXE) {
            ents[nE].v = (float)(pw * v); ents[nE].d = d; ++nE; ++cnt;
          }
        }
        if (cnt && nC < MAXC) {
          cells[nC].k = kOfPathA(p, a); cells[nC].n = nOffT[io] + c;
          cells[nC].off = off; cells[nC].cnt = cnt; ++nC;
        }
      }
  }
}

// ---------------- launch ----------------
extern "C" void kernel_launch(void* const* d_in, const int* in_sizes, int n_in,
                              void* d_out, int out_size, void* d_ws, size_t ws_size,
                              hipStream_t stream) {
  const float* x1  = (const float*)d_in[0];
  const float* x2  = (const float*)d_in[1];
  const float* wgt = (const float*)d_in[2];
  float* out = (float*)d_out;
  const int nSamples = in_sizes[0] / DIM1;

  if (ws_size < WS_TOTAL) return;

  // Static host storage so graph replays re-read valid memory; contents are
  // recomputed identically on every call (deterministic, no guards/caching).
  static Cell hCells[MAXC];
  static Ent  hEnts[MAXE];
  int nC = 0, nE = 0;
  build_tables(hCells, nC, hEnts, nE);

  char* ws = (char*)d_ws;
  hipMemcpyAsync(ws + WS_CELLS, hCells, (size_t)nC * 16, hipMemcpyHostToDevice, stream);
  hipMemcpyAsync(ws + WS_ENTS,  hEnts,  (size_t)nE * 8,  hipMemcpyHostToDevice, stream);

  hipFuncSetAttribute(reinterpret_cast<const void*>(tp_uvu_wmma_kernel),
                      hipFuncAttributeMaxDynamicSharedMemorySize, SMEM_TOTAL);

  int grid = (nSamples + WAVES - 1) / WAVES;
  if (grid > 2048) grid = 2048;              // persistent waves, loop over samples
  tp_uvu_wmma_kernel<<<grid, BLOCK, SMEM_TOTAL, stream>>>(
      x1, x2, wgt, out,
      (const int4*)(ws + WS_CELLS), nC,
      (const Ent*)(ws + WS_ENTS), nSamples);
}
